// GRIL_24008867184895
// MI455X (gfx1250) — compile-verified, matrix operand order checked by reading, not run on hardware
//
#include <hip/hip_runtime.h>

// ---------------------------------------------------------------------------
// GRIL on MI455X (gfx1250): bf16 WMMA graph diffusion + fp32 pixel convs.
// B=8, N=1024, T=24, HID=64. Diffusion = [M,1024] x [1024,1024]^T matmuls,
// M=512 (gc) or 528 (gates/candidate). All data lives in L2 (192MB).
// Diffusion kernel v3: adjacency tiles staged to LDS via the Tensor Data
// Mover (double-buffered, TENSORcnt-synced), X fragments double-buffered in
// registers, and a fully convergent inner loop (partial waves clamped onto
// the last m-tile instead of predicated out) so no exec-mask churn or
// accumulator phi-copies surround the WMMA chain.
// ---------------------------------------------------------------------------

#define B_   8
#define N_   1024
#define T_   24
#define HID_ 64
#define C66_ 66
#define MGC_ (B_ * HID_)   // 512
#define MXH_ (B_ * C66_)   // 528
#define ZBLK_ (MXH_ * N_)  // elements per diffusion block

typedef __attribute__((ext_vector_type(16))) __bf16 v16bf;
typedef __attribute__((ext_vector_type(8)))  float  v8f;
typedef unsigned int __attribute__((ext_vector_type(4))) u32x4;
typedef int          __attribute__((ext_vector_type(4))) i32x4;
typedef int          __attribute__((ext_vector_type(8))) i32x8;

#define HAS_TDM __has_builtin(__builtin_amdgcn_tensor_load_to_lds)

__device__ __forceinline__ unsigned short f2bf(float f) {
    union { float f; unsigned u; } v; v.f = f;
    unsigned r = v.u + 0x7FFFu + ((v.u >> 16) & 1u);   // round to nearest even
    return (unsigned short)(r >> 16);
}
__device__ __forceinline__ float bf2f(unsigned short h) {
    union { unsigned u; float f; } v; v.u = ((unsigned)h) << 16;
    return v.f;
}

// Load one WMMA 16-bit fragment share for this lane: 8 bf16 at p (K-low oct)
// and 8 bf16 at p+16 (K-high oct). Both 16B aligned. Works on global or LDS.
__device__ __forceinline__ v16bf load_frag(const unsigned short* p) {
    union { uint4 u[2]; v16bf v; } t;
    t.u[0] = *(const uint4*)(p);
    t.u[1] = *(const uint4*)(p + 16);
    return t.v;
}

#if HAS_TDM
// Issue a TDM 2D tile load: 64 rows x 32 bf16 elements from a row-major
// [.,1024] bf16 tensor into LDS at lds_off (rows packed at 64B pitch).
__device__ __forceinline__ void stage_tile_tdm(unsigned lds_off,
                                               const unsigned short* gsrc) {
    unsigned long ga = (unsigned long)gsrc;
    u32x4 g0;
    g0[0] = 1u;                                   // count=1, user descriptor
    g0[1] = lds_off;                              // D#.lds_addr
    g0[2] = (unsigned)(ga & 0xFFFFFFFFu);         // global_addr[31:0]
    g0[3] = (unsigned)((ga >> 32) & 0x01FFFFFFu)  // global_addr[56:32]
          | (2u << 30);                           // type=2 ("image")
    i32x8 g1;
    g1[0] = (int)(1u << 16);                      // data_size=1 (2B), mask=0
    g1[1] = (int)((1024u & 0xFFFFu) << 16);       // tensor_dim0=1024 (lo16)
    g1[2] = (int)((1024u & 0xFFFFu) << 16);       // td0 hi=0 | tensor_dim1 lo16
    g1[3] = (int)(32u << 16);                     // td1 hi=0 | tile_dim0=32
    g1[4] = (int)(64u);                           // tile_dim1=64, tile_dim2=0
    g1[5] = (int)(1024u);                         // tensor_dim0_stride lo32
    g1[6] = 0;                                    // st0 hi | st1 lo
    g1[7] = 0;
    i32x4 z4 = {0, 0, 0, 0};
#if defined(__clang_major__) && (__clang_major__ >= 23)
    i32x8 z8 = {0, 0, 0, 0, 0, 0, 0, 0};
    __builtin_amdgcn_tensor_load_to_lds(g0, g1, z4, z4, z8, 0);
#else
    __builtin_amdgcn_tensor_load_to_lds(g0, g1, z4, z4, 0);
#endif
}
#endif

// ---------------------------------------------------------------------------
// Diffusion: O[s][m][w] = sum_v X[s][m][v] * A[s][w][v]   (bf16 in/out, fp32
// accumulate). Block = 8 waves x (16 rows) over a shared 64-column strip;
// the 64x32 adjacency tile is staged once per k-step into LDS (TDM when
// available), shared by all 8 waves. X fragments double-buffered in VGPRs.
// M must be a multiple of 16; waves past the last m-tile are clamped onto it
// (they redundantly recompute/store identical data) so the block stays fully
// convergent: every wave runs the WMMA chain with EXEC all ones.
// ---------------------------------------------------------------------------
__global__ __launch_bounds__(256) void diffuse_wmma_kernel(
    const unsigned short* __restrict__ X, long xBlockStride,
    const unsigned short* __restrict__ Abf,
    unsigned short* __restrict__ Out, long outBlockStride, int M)
{
    __shared__ unsigned short sbuf[2][64 * 32];   // 2 x 4KB B-tile buffers

    const int s = blockIdx.z;
    const unsigned short* Xs = X   + (long)s * xBlockStride;
    const unsigned short* As = Abf + (long)s * (long)(N_ * N_);
    unsigned short*       Os = Out + (long)s * outBlockStride;

    const int wave   = threadIdx.x >> 5;
    const int lane   = threadIdx.x & 31;
    const int mtiles = M >> 4;
    int mtile = blockIdx.x * 8 + wave;
    mtile = (mtile < mtiles) ? mtile : (mtiles - 1);   // clamp: stay convergent
    const int m0   = mtile * 16;
    const int w0   = blockIdx.y * 64;
    const int lrow = lane & 15;
    const int hi8  = ((lane >> 4) & 1) * 8;

    const unsigned short* xrow  = Xs + (long)(m0 + lrow) * N_ + hi8;
    const unsigned short* atile = As + (long)w0 * N_;   // tile origin row w0

    v8f acc0 = {}; v8f acc1 = {}; v8f acc2 = {}; v8f acc3 = {};

    // ---- prologue: stage k0 = 0 tile ----
#if HAS_TDM
    if (wave == 0)
        stage_tile_tdm((unsigned)(unsigned long)&sbuf[0][0], atile);
    __builtin_amdgcn_s_wait_tensorcnt(0);   // no-op for non-issuing waves
#else
    {
        const int row = threadIdx.x >> 2, chunk = threadIdx.x & 3;
        *(uint4*)(&sbuf[0][row * 32 + chunk * 8]) =
            *(const uint4*)(atile + (long)row * N_ + chunk * 8);
    }
#endif
    __syncthreads();

    v16bf a_cur = load_frag(xrow);

    const int kIter = N_ / 32;
    for (int it = 0; it < kIter; ++it) {
        const int cur = it & 1, nxt = cur ^ 1;
        const int k0n = (it + 1) * 32;
        const int kld = (it + 1 < kIter) ? k0n : 0;   // clamp last prefetch

        // stage next tile (async under TDM) while computing on current one
        if (it + 1 < kIter) {
#if HAS_TDM
            if (wave == 0)
                stage_tile_tdm((unsigned)(unsigned long)&sbuf[nxt][0],
                               atile + k0n);
#else
            const int row = threadIdx.x >> 2, chunk = threadIdx.x & 3;
            *(uint4*)(&sbuf[nxt][row * 32 + chunk * 8]) =
                *(const uint4*)(atile + (long)row * N_ + k0n + chunk * 8);
#endif
        }

        // prefetch next X fragment into registers (overlaps with WMMA)
        __builtin_prefetch(xrow + kld + 32, 0, 2);
        v16bf a_nxt = load_frag(xrow + kld);

        const unsigned short* bb = &sbuf[cur][0];
        v16bf b0 = load_frag(bb + ( 0 + lrow) * 32 + hi8);
        v16bf b1 = load_frag(bb + (16 + lrow) * 32 + hi8);
        v16bf b2 = load_frag(bb + (32 + lrow) * 32 + hi8);
        v16bf b3 = load_frag(bb + (48 + lrow) * 32 + hi8);
        acc0 = __builtin_amdgcn_wmma_f32_16x16x32_bf16(false, a_cur, false, b0, (short)0, acc0, false, false);
        acc1 = __builtin_amdgcn_wmma_f32_16x16x32_bf16(false, a_cur, false, b1, (short)0, acc1, false, false);
        acc2 = __builtin_amdgcn_wmma_f32_16x16x32_bf16(false, a_cur, false, b2, (short)0, acc2, false, false);
        acc3 = __builtin_amdgcn_wmma_f32_16x16x32_bf16(false, a_cur, false, b3, (short)0, acc3, false, false);
        a_cur = a_nxt;

#if HAS_TDM
        __builtin_amdgcn_s_wait_tensorcnt(0);   // uniform: no saveexec needed
#endif
        __syncthreads();   // publish staged tile; all reads of 'cur' are done
    }

    #pragma unroll
    for (int r = 0; r < 8; ++r) {
        const long mr = (long)(m0 + r + hi8) * N_;
        Os[mr + w0 +  0 + lrow] = f2bf(acc0[r]);
        Os[mr + w0 + 16 + lrow] = f2bf(acc1[r]);
        Os[mr + w0 + 32 + lrow] = f2bf(acc2[r]);
        Os[mr + w0 + 48 + lrow] = f2bf(acc3[r]);
    }
}

// ---------------------------------------------------------------------------
// Init: adjacency fp32 -> bf16; broadcast h0 to all batches.
// ---------------------------------------------------------------------------
__global__ void cvt_adj_kernel(const float* __restrict__ adj,
                               unsigned short* __restrict__ Abf) {
    long i = (long)blockIdx.x * 256 + threadIdx.x;
    if (i < (long)2 * N_ * N_) Abf[i] = f2bf(adj[i]);
}
__global__ void init_h_kernel(const float* __restrict__ h0,
                              float* __restrict__ h) {
    long i = (long)blockIdx.x * 256 + threadIdx.x;     // over B*HID*N
    if (i < (long)B_ * HID_ * N_) h[i] = h0[i % (HID_ * N_)];
}

// ---------------------------------------------------------------------------
// K1: xs_hat_1 = fs_W . h ; x1 = sel(mask, x, xs_hat_1);
//     x_in = li_W @ [x1, m, h] -> bf16 buffer for diffusion. One pixel/block.
// ---------------------------------------------------------------------------
__global__ __launch_bounds__(64) void pre_kernel(
    const float* __restrict__ x, const float* __restrict__ mask,
    const float* __restrict__ h,
    const float* __restrict__ fs_W, const float* __restrict__ fs_b,
    const float* __restrict__ li_W, const float* __restrict__ li_b,
    unsigned short* __restrict__ Xgc, float* __restrict__ x1buf,
    float* __restrict__ out_pred, int t)
{
    __shared__ float sh_h[HID_];
    __shared__ float sh_red[HID_];
    __shared__ float sh_x1m[2];
    const int pix = blockIdx.x, b = pix >> 10, n = pix & (N_ - 1);
    const int c = threadIdx.x;
    const float hv = h[((long)b * HID_ + c) * N_ + n];
    sh_h[c] = hv;
    sh_red[c] = fs_W[c] * hv;
    __syncthreads();
    for (int st = 32; st > 0; st >>= 1) {
        if (c < st) sh_red[c] += sh_red[c + st];
        __syncthreads();
    }
    if (c == 0) {
        const long pidx = ((long)b * N_ + n) * T_ + t;
        const float xhat1 = sh_red[0] + fs_b[0];
        out_pred[pidx] = xhat1;
        const float m  = mask[pidx];
        const float x1 = (m > 0.f) ? x[pidx] : xhat1;
        x1buf[(long)b * N_ + n] = x1;
        sh_x1m[0] = x1; sh_x1m[1] = m;
    }
    __syncthreads();
    const float* w = li_W + c * 66;
    float acc = li_b[c] + w[0] * sh_x1m[0] + w[1] * sh_x1m[1];
    #pragma unroll 8
    for (int j = 0; j < HID_; ++j) acc += w[2 + j] * sh_h[j];
    Xgc[((long)b * HID_ + c) * N_ + n] = f2bf(acc);
}

// ---------------------------------------------------------------------------
// K3: gc conv on diffusion output, lo conv, PReLU, repr out, xs_hat_2,
//     x2 select, build xh = [x2, m, h] bf16 (Zxh block 0).
// ---------------------------------------------------------------------------
__global__ __launch_bounds__(64) void post_gc_kernel(
    const float* __restrict__ x, const float* __restrict__ mask,
    const unsigned short* __restrict__ Ygc, const float* __restrict__ h,
    const float* __restrict__ gc_W, const float* __restrict__ gc_b,
    const float* __restrict__ lo_W, const float* __restrict__ lo_b,
    const float* __restrict__ ro_W, const float* __restrict__ ro_b,
    const float* __restrict__ prelu_a,
    const float* __restrict__ x1buf, float* __restrict__ x2buf,
    unsigned short* __restrict__ Zxh,
    float* __restrict__ out_repr, float* __restrict__ out_imp, int t)
{
    __shared__ float sh_y[2 * HID_];
    __shared__ float sh_h[HID_];
    __shared__ float sh_o[HID_];
    __shared__ float sh_r[HID_];
    const int pix = blockIdx.x, b = pix >> 10, n = pix & (N_ - 1);
    const int c = threadIdx.x;
    sh_y[c]        = bf2f(Ygc[((long)0 * MGC_ + b * HID_ + c) * N_ + n]);
    sh_y[HID_ + c] = bf2f(Ygc[((long)1 * MGC_ + b * HID_ + c) * N_ + n]);
    const float hv = h[((long)b * HID_ + c) * N_ + n];
    sh_h[c] = hv;
    __syncthreads();
    {
        const float* w = gc_W + c * 128;
        float acc = gc_b[c];
        for (int i = 0; i < 128; ++i) acc += w[i] * sh_y[i];
        sh_o[c] = acc;
    }
    __syncthreads();
    const float* w2 = lo_W + c * 128;
    float acc2 = lo_b[c];
    for (int i = 0; i < HID_; ++i) acc2 += w2[i] * sh_o[i];
    for (int i = 0; i < HID_; ++i) acc2 += w2[HID_ + i] * sh_h[i];
    const float a = prelu_a[0];
    const float out2 = (acc2 >= 0.f) ? acc2 : a * acc2;
    out_repr[(((long)b * 128 + c) * N_ + n) * T_ + t]        = out2;
    out_repr[(((long)b * 128 + HID_ + c) * N_ + n) * T_ + t] = hv;
    sh_r[c] = ro_W[c] * out2 + ro_W[HID_ + c] * hv;
    __syncthreads();
    for (int st = 32; st > 0; st >>= 1) {
        if (c < st) sh_r[c] += sh_r[c + st];
        __syncthreads();
    }
    if (c == 0) {
        const long pidx = ((long)b * N_ + n) * T_ + t;
        const float xhat2 = sh_r[0] + ro_b[0];
        out_imp[pidx] = xhat2;
        const float m  = mask[pidx];
        const float x2 = (m > 0.f) ? x1buf[(long)b * N_ + n] : xhat2;
        x2buf[(long)b * N_ + n] = x2;
        Zxh[((long)b * C66_ + 0) * N_ + n] = f2bf(x2);
        Zxh[((long)b * C66_ + 1) * N_ + n] = f2bf(m);
    }
    Zxh[((long)b * C66_ + 2 + c) * N_ + n] = f2bf(hv);
}

// ---------------------------------------------------------------------------
// K5: r,u gates from [xh, A f xh, A2 f xh, A b xh, A2 b xh]; build
//     xc = [x2, m, r*h] (Zxc block 0); store u.
// ---------------------------------------------------------------------------
__global__ __launch_bounds__(64) void gates_kernel(
    const unsigned short* __restrict__ Zxh, const float* __restrict__ h,
    const float* __restrict__ f_W, const float* __restrict__ f_b,
    const float* __restrict__ u_W, const float* __restrict__ u_b,
    const float* __restrict__ x2buf, const float* __restrict__ mask,
    float* __restrict__ ubuf, unsigned short* __restrict__ Zxc, int t)
{
    __shared__ float sz[5 * C66_];
    const int pix = blockIdx.x, b = pix >> 10, n = pix & (N_ - 1);
    const int c = threadIdx.x;
    for (int i = c; i < 5 * C66_; i += 64) {
        const int blk = i / C66_, ch = i % C66_;
        sz[i] = bf2f(Zxh[((long)blk * MXH_ + b * C66_ + ch) * N_ + n]);
    }
    __syncthreads();
    const float* fw = f_W + c * 330;
    const float* uw = u_W + c * 330;
    float fa = f_b[c], ua = u_b[c];
    for (int i = 0; i < 330; ++i) { const float z = sz[i]; fa += fw[i] * z; ua += uw[i] * z; }
    const float r = 1.f / (1.f + __expf(-fa));
    const float u = 1.f / (1.f + __expf(-ua));
    ubuf[((long)b * HID_ + c) * N_ + n] = u;
    const float hv = h[((long)b * HID_ + c) * N_ + n];
    Zxc[((long)b * C66_ + 2 + c) * N_ + n] = f2bf(r * hv);
    if (c == 0) {
        Zxc[((long)b * C66_ + 0) * N_ + n] = f2bf(x2buf[(long)b * N_ + n]);
        Zxc[((long)b * C66_ + 1) * N_ + n] = f2bf(mask[((long)b * N_ + n) * T_ + t]);
    }
}

// K8: candidate + GRU update (pre-norm h_new).
__global__ __launch_bounds__(64) void cand_kernel(
    const unsigned short* __restrict__ Zxc,
    const float* __restrict__ c_W, const float* __restrict__ c_b,
    const float* __restrict__ ubuf, const float* __restrict__ h,
    float* __restrict__ hnew)
{
    __shared__ float sz[5 * C66_];
    const int pix = blockIdx.x, b = pix >> 10, n = pix & (N_ - 1);
    const int c = threadIdx.x;
    for (int i = c; i < 5 * C66_; i += 64) {
        const int blk = i / C66_, ch = i % C66_;
        sz[i] = bf2f(Zxc[((long)blk * MXH_ + b * C66_ + ch) * N_ + n]);
    }
    __syncthreads();
    const float* cw = c_W + c * 330;
    float ca = c_b[c];
    for (int i = 0; i < 330; ++i) ca += cw[i] * sz[i];
    const float cv = tanhf(ca);
    const long idx = ((long)b * HID_ + c) * N_ + n;
    const float u = ubuf[idx];
    hnew[idx] = u * h[idx] + (1.f - u) * cv;
}

// K9: GroupNorm(1 group) over (C,N) per sample; writes h and states output.
__global__ __launch_bounds__(1024) void gnorm_kernel(
    const float* __restrict__ hnew,
    const float* __restrict__ gn_g, const float* __restrict__ gn_beta,
    float* __restrict__ h, float* __restrict__ out_states, int t)
{
    __shared__ float s1[N_];
    __shared__ float s2[N_];
    const int b = blockIdx.x, n = threadIdx.x;
    float sum = 0.f, sq = 0.f;
    for (int c = 0; c < HID_; ++c) {
        const float v = hnew[((long)b * HID_ + c) * N_ + n];
        sum += v; sq += v * v;
    }
    s1[n] = sum; s2[n] = sq;
    __syncthreads();
    for (int st = 512; st > 0; st >>= 1) {
        if (n < st) { s1[n] += s1[n + st]; s2[n] += s2[n + st]; }
        __syncthreads();
    }
    const float inv = 1.f / (float)(HID_ * N_);
    const float mu  = s1[0] * inv;
    const float var = s2[0] * inv - mu * mu;
    const float rstd = rsqrtf(var + 1e-5f);
    for (int c = 0; c < HID_; ++c) {
        const long idx = ((long)b * HID_ + c) * N_ + n;
        const float o = (hnew[idx] - mu) * rstd * gn_g[c] + gn_beta[c];
        h[idx] = o;
        out_states[idx * T_ + t] = o;
    }
}

// ---------------------------------------------------------------------------
extern "C" void kernel_launch(void* const* d_in, const int* in_sizes, int n_in,
                              void* d_out, int out_size, void* d_ws, size_t ws_size,
                              hipStream_t stream) {
    const float* x     = (const float*)d_in[0];
    const float* mask  = (const float*)d_in[1];
    const float* adj   = (const float*)d_in[2];
    const float* h0    = (const float*)d_in[3];
    const float* fs_W  = (const float*)d_in[4];
    const float* fs_b  = (const float*)d_in[5];
    const float* li_W  = (const float*)d_in[6];
    const float* li_b  = (const float*)d_in[7];
    const float* gc_W  = (const float*)d_in[8];
    const float* gc_b  = (const float*)d_in[9];
    const float* lo_W  = (const float*)d_in[10];
    const float* lo_b  = (const float*)d_in[11];
    const float* ro_W  = (const float*)d_in[12];
    const float* ro_b  = (const float*)d_in[13];
    const float* prelu = (const float*)d_in[14];
    const float* f_W   = (const float*)d_in[15];
    const float* f_b   = (const float*)d_in[16];
    const float* u_W   = (const float*)d_in[17];
    const float* u_b   = (const float*)d_in[18];
    const float* c_W   = (const float*)d_in[19];
    const float* c_b   = (const float*)d_in[20];
    const float* gn_g  = (const float*)d_in[21];
    const float* gn_b  = (const float*)d_in[22];
    (void)in_sizes; (void)n_in; (void)out_size;

    // d_out sections: imputations | predictions | representations | states
    float* out_imp    = (float*)d_out;
    float* out_pred   = out_imp  + (long)B_ * N_ * T_;
    float* out_repr   = out_pred + (long)B_ * N_ * T_;
    float* out_states = out_repr + (long)B_ * 128 * N_ * T_;

    // Workspace layout (bytes)
    char* ws = (char*)d_ws;
    const size_t NEED = 24510464;
    if (ws_size < NEED) return;
    unsigned short* Abf  = (unsigned short*)(ws + 0);          // 2*1024*1024 bf16
    float*          hbuf = (float*)(ws + 4194304);             // B*64*N
    float*          hnew = (float*)(ws + 6291456);
    float*          ubuf = (float*)(ws + 8388608);
    unsigned short* Xgc  = (unsigned short*)(ws + 10485760);   // [512][1024] bf16
    unsigned short* Ygc  = (unsigned short*)(ws + 11534336);   // [2][512][1024] bf16
    unsigned short* Zxh  = (unsigned short*)(ws + 13631488);   // [5][528][1024] bf16
    unsigned short* Zxc  = (unsigned short*)(ws + 19038208);   // [5][528][1024] bf16
    float*          x1b  = (float*)(ws + 24444928);            // [B][N]
    float*          x2b  = (float*)(ws + 24477696);            // [B][N]

    cvt_adj_kernel<<<(2 * N_ * N_) / 256, 256, 0, stream>>>(adj, Abf);
    init_h_kernel<<<(B_ * HID_ * N_) / 256, 256, 0, stream>>>(h0, hbuf);

    const long ZB = (long)ZBLK_;
    for (int t = 0; t < T_; ++t) {
        pre_kernel<<<B_ * N_, 64, 0, stream>>>(
            x, mask, hbuf, fs_W, fs_b, li_W, li_b, Xgc, x1b, out_pred, t);

        // order-1 gconv diffusion: Ygc[s] = Xgc @ A_s^T  (M=512)
        diffuse_wmma_kernel<<<dim3(4, 16, 2), 256, 0, stream>>>(
            Xgc, 0L, Abf, Ygc, (long)MGC_ * N_, MGC_);

        post_gc_kernel<<<B_ * N_, 64, 0, stream>>>(
            x, mask, Ygc, hbuf, gc_W, gc_b, lo_W, lo_b, ro_W, ro_b, prelu,
            x1b, x2b, Zxh, out_repr, out_imp, t);

        // order-2 gate diffusion on xh (M=528): blocks 1,3 then 2,4
        diffuse_wmma_kernel<<<dim3(5, 16, 2), 256, 0, stream>>>(
            Zxh, 0L, Abf, Zxh + 1 * ZB, 2 * ZB, MXH_);
        diffuse_wmma_kernel<<<dim3(5, 16, 2), 256, 0, stream>>>(
            Zxh + 1 * ZB, 2 * ZB, Abf, Zxh + 2 * ZB, 2 * ZB, MXH_);

        gates_kernel<<<B_ * N_, 64, 0, stream>>>(
            Zxh, hbuf, f_W, f_b, u_W, u_b, x2b, mask, ubuf, Zxc, t);

        // order-2 candidate diffusion on xc
        diffuse_wmma_kernel<<<dim3(5, 16, 2), 256, 0, stream>>>(
            Zxc, 0L, Abf, Zxc + 1 * ZB, 2 * ZB, MXH_);
        diffuse_wmma_kernel<<<dim3(5, 16, 2), 256, 0, stream>>>(
            Zxc + 1 * ZB, 2 * ZB, Abf, Zxc + 2 * ZB, 2 * ZB, MXH_);

        cand_kernel<<<B_ * N_, 64, 0, stream>>>(Zxc, c_W, c_b, ubuf, hbuf, hnew);

        gnorm_kernel<<<B_, 1024, 0, stream>>>(hnew, gn_g, gn_b, hbuf, out_states, t);
    }
}
